// VectorQuantizer_17325898072130
// MI455X (gfx1250) — compile-verified, matrix operand order checked by reading, not run on hardware
//
#include <hip/hip_runtime.h>
#include <stdint.h>

// ---------------------------------------------------------------------------
// VQ-VAE vector quantizer for MI455X (gfx1250, wave32, WMMA bf16 path)
//   - distance GEMM on v_wmma_f32_16x16x32_bf16
//   - double-buffered LDS codebook tiles staged via async global->LDS copies
// ---------------------------------------------------------------------------

typedef __attribute__((ext_vector_type(16))) __bf16        v16bf;
typedef __attribute__((ext_vector_type(8)))  float         v8f;
typedef __attribute__((ext_vector_type(8)))  unsigned int  v8u;

// async-copy builtin operand types (per hipcc diagnostic: v4i in AS1/AS3)
typedef int v4i __attribute__((vector_size(16)));
typedef __attribute__((address_space(1))) v4i gv4i;   // global
typedef __attribute__((address_space(3))) v4i lv4i;   // LDS

static constexpr int Bn = 32, Cc = 256, Hh = 32, Ww = 32;
static constexpr int HW = Hh * Ww;          // 1024
static constexpr int Np = Bn * HW;          // 32768 pixels
static constexpr int Kc = 1024;             // codes
static constexpr int Dd = 256;              // embedding dim
static constexpr int CODE_TILE  = 32;       // codes per LDS tile (double buffered)
static constexpr int NT         = Kc / CODE_TILE;   // 32 tiles
static constexpr int LDS_STRIDE = 132;      // dwords per code row (128 + 4 pad)

#if defined(__has_builtin)
# if __has_builtin(__builtin_amdgcn_global_load_async_to_lds_b128)
#  define HAVE_ASYNC_COPY 1
# endif
#endif

__device__ __forceinline__ uint32_t bf16_rne(float f) {
    uint32_t u = __float_as_uint(f);
    return (u + 0x7FFFu + ((u >> 16) & 1u)) >> 16;
}
__device__ __forceinline__ uint32_t pack_bf16(float lo, float hi) {
    return bf16_rne(lo) | (bf16_rne(hi) << 16);
}

__device__ __forceinline__ void async_wait_all() {
#if defined(HAVE_ASYNC_COPY)
# if __has_builtin(__builtin_amdgcn_s_wait_asynccnt)
    __builtin_amdgcn_s_wait_asynccnt(0);
# else
    asm volatile("s_wait_asynccnt 0x0" ::: "memory");
# endif
#endif
}

// ---------------------------------------------------------------------------
// Kernel 0: convert embedding f32 -> packed bf16 (d_ws) + per-code ||e||^2
// ---------------------------------------------------------------------------
__global__ __launch_bounds__(64) void vq_prep(const float* __restrict__ emb,
                                              uint32_t* __restrict__ ebf,
                                              float* __restrict__ enorm) {
    const int k = blockIdx.x;       // code 0..1023
    const int t = threadIdx.x;      // 0..63, 4 channels each
    const float4 v = ((const float4*)(emb + (size_t)k * Dd))[t];
    ebf[(size_t)k * 128 + t * 2 + 0] = pack_bf16(v.x, v.y);
    ebf[(size_t)k * 128 + t * 2 + 1] = pack_bf16(v.z, v.w);
    __shared__ float red[64];
    red[t] = v.x * v.x + v.y * v.y + v.z * v.z + v.w * v.w;
    __syncthreads();
    for (int off = 32; off > 0; off >>= 1) {
        if (t < off) red[t] += red[t + off];
        __syncthreads();
    }
    if (t == 0) enorm[k] = red[0];
}

// ---------------------------------------------------------------------------
// Stage one 32-code tile (16 KB) into an LDS buffer.
//   256 threads: 8 threads per code, 4 x b128 each. Async when available.
// ---------------------------------------------------------------------------
__device__ __forceinline__ void stage_tile(const uint32_t* __restrict__ ebf,
                                           const float* __restrict__ enorm,
                                           uint32_t* lds_tile, float* lds_norm,
                                           int tile, int tid) {
    const int code = tid >> 3;      // 0..31
    const int part = tid & 7;       // 0..7, 16 dwords each
    const uint32_t* src = ebf + (size_t)(tile * CODE_TILE + code) * 128 + part * 16;
    uint32_t* dst = lds_tile + code * LDS_STRIDE + part * 16;
#if defined(HAVE_ASYNC_COPY)
#pragma unroll
    for (int j = 0; j < 4; ++j) {
        __builtin_amdgcn_global_load_async_to_lds_b128(
            (gv4i*)(uintptr_t)(src + j * 4),
            (lv4i*)(uint32_t)(uintptr_t)(dst + j * 4),
            0, 0);
    }
#else
#pragma unroll
    for (int j = 0; j < 4; ++j)
        *(uint4*)(dst + j * 4) = *(const uint4*)(src + j * 4);
#endif
    if (tid < CODE_TILE) lds_norm[tid] = enorm[tile * CODE_TILE + tid];
}

// ---------------------------------------------------------------------------
// Kernel 1: distance GEMM via v_wmma_f32_16x16x32_bf16 + per-row argmin
//   block = 256 threads = 8 waves; wave owns 16 pixel rows (A tile in regs)
// ---------------------------------------------------------------------------
__global__ __launch_bounds__(256) void vq_argmin(const float* __restrict__ z,
                                                 const uint32_t* __restrict__ ebf,
                                                 const float* __restrict__ enorm,
                                                 int* __restrict__ idx_out) {
    __shared__ uint32_t lds_e[2][CODE_TILE * LDS_STRIDE];  // 2 x ~16.5 KB
    __shared__ float    lds_en[2][CODE_TILE];

    const int tid   = threadIdx.x;
    const int wave  = tid >> 5;
    const int lane  = tid & 31;
    const int mh    = lane & 15;    // A: row-in-tile / B: column (code)
    const int khalf = lane >> 4;    // K split between lane halves

    // Kick off async staging of tile 0 first so it overlaps the A preload.
    stage_tile(ebf, enorm, lds_e[0], lds_en[0], 0, tid);

    const int rowbase = blockIdx.x * 128 + wave * 16;
    const int row = rowbase + mh;                 // this lane's A row (pixel)
    const int pb  = row >> 10;                    // image in batch
    const int phw = row & (HW - 1);               // spatial offset
    const float* zrow = z + (size_t)pb * (Cc * HW) + phw;  // elem k at zrow[k*HW]

    // --- Preload A fragments (ISA 16-bit A 16x32 layout), K = 8 steps of 32
    uint32_t afr[8][8];
#pragma unroll
    for (int kk = 0; kk < 8; ++kk) {
#pragma unroll
        for (int j = 0; j < 8; ++j) {
            const int k0 = kk * 32 + ((j < 4) ? 0 : 16) + khalf * 8 + 2 * (j & 3);
            const float lo = zrow[(size_t)k0 * HW];
            const float hi = zrow[(size_t)(k0 + 1) * HW];
            afr[kk][j] = pack_bf16(lo, hi);
        }
    }

    float bestd[8];
    int   besti[8];
#pragma unroll
    for (int r = 0; r < 8; ++r) { bestd[r] = 3.4e38f; besti[r] = 0; }

    async_wait_all();
    __syncthreads();

    for (int ct = 0; ct < NT; ++ct) {
        const int cur = ct & 1;
        if (ct + 1 < NT)
            stage_tile(ebf, enorm, lds_e[cur ^ 1], lds_en[cur ^ 1], ct + 1, tid);

#pragma unroll
        for (int t = 0; t < CODE_TILE / 16; ++t) {   // 16-code subtiles
            const uint32_t* brow = &lds_e[cur][(t * 16 + mh) * LDS_STRIDE + khalf * 4];
            // software-pipeline the B fragments so WMMA overlaps ds_load
            uint4 b0 = *(const uint4*)(brow);
            uint4 b1 = *(const uint4*)(brow + 8);
            v8f acc = {};
#pragma unroll
            for (int kk = 0; kk < 8; ++kk) {
                uint4 n0 = b0, n1 = b1;
                if (kk < 7) {
                    n0 = *(const uint4*)(brow + (kk + 1) * 16);
                    n1 = *(const uint4*)(brow + (kk + 1) * 16 + 8);
                }
                v8u bu; bu[0]=b0.x; bu[1]=b0.y; bu[2]=b0.z; bu[3]=b0.w;
                        bu[4]=b1.x; bu[5]=b1.y; bu[6]=b1.z; bu[7]=b1.w;
                v8u au;
#pragma unroll
                for (int j = 0; j < 8; ++j) au[j] = afr[kk][j];
                acc = __builtin_amdgcn_wmma_f32_16x16x32_bf16(
                        false, __builtin_bit_cast(v16bf, au),
                        false, __builtin_bit_cast(v16bf, bu),
                        (short)0, acc, false, false);
                b0 = n0; b1 = n1;
            }
            const int   code = ct * CODE_TILE + t * 16 + mh;   // lane's column
            const float en   = lds_en[cur][t * 16 + mh];
#pragma unroll
            for (int r = 0; r < 8; ++r) {
                // D layout: lane l, vgpr r -> row (r + 8*khalf), col (l&15)
                const float dist = en - 2.0f * acc[r];
                if (dist < bestd[r]) { bestd[r] = dist; besti[r] = code; }
            }
        }

        async_wait_all();   // our own async copies for tile ct+1 done
        __syncthreads();    // all waves' copies visible + LDS reads retired
    }

    // min-reduce across the 16 columns within each lane half (ties -> lowest idx)
#pragma unroll
    for (int off = 1; off < 16; off <<= 1) {
#pragma unroll
        for (int r = 0; r < 8; ++r) {
            const float od = __shfl_xor(bestd[r], off, 32);
            const int   oi = __shfl_xor(besti[r], off, 32);
            if (od < bestd[r] || (od == bestd[r] && oi < besti[r])) {
                bestd[r] = od; besti[r] = oi;
            }
        }
    }
    if (mh == 0) {
        const int mbase = rowbase + khalf * 8;
#pragma unroll
        for (int r = 0; r < 8; ++r) idx_out[mbase + r] = besti[r];
    }
}

// ---------------------------------------------------------------------------
// Kernel 2: gather z_q = E[idx] into (B,C,H,W) + per-block sum of (z_q - z)^2
// ---------------------------------------------------------------------------
__global__ __launch_bounds__(256) void vq_gather(const float* __restrict__ z,
                                                 const float* __restrict__ emb,
                                                 const int* __restrict__ idx,
                                                 float* __restrict__ zq_out,
                                                 float* __restrict__ partials) {
    const int n = blockIdx.x * 256 + threadIdx.x;   // pixel
    const int c = blockIdx.y;                       // channel
    const int ii = idx[n];
    const float zq = emb[(size_t)ii * Dd + c];
    const int pb = n >> 10, phw = n & (HW - 1);
    const size_t off = (size_t)pb * (Cc * HW) + (size_t)c * HW + phw;
    const float diff = zq - z[off];
    zq_out[off] = zq;

    __shared__ float red[256];
    red[threadIdx.x] = diff * diff;
    __syncthreads();
    for (int o = 128; o > 0; o >>= 1) {
        if (threadIdx.x < o) red[threadIdx.x] += red[threadIdx.x + o];
        __syncthreads();
    }
    if (threadIdx.x == 0)
        partials[(size_t)blockIdx.y * gridDim.x + blockIdx.x] = red[0];
}

// ---------------------------------------------------------------------------
// Kernel 3: deterministic final reduction -> vq_loss = 1.25 * mean(diff^2)
// ---------------------------------------------------------------------------
__global__ __launch_bounds__(256) void vq_loss_reduce(const float* __restrict__ partials,
                                                      float* __restrict__ loss) {
    __shared__ float red[256];
    float s = 0.0f;
    for (int i = threadIdx.x; i < 32768; i += 256) s += partials[i];
    red[threadIdx.x] = s;
    __syncthreads();
    for (int o = 128; o > 0; o >>= 1) {
        if (threadIdx.x < o) red[threadIdx.x] += red[threadIdx.x + o];
        __syncthreads();
    }
    if (threadIdx.x == 0)
        *loss = 1.25f * red[0] / (float)(Np * Dd);
}

// ---------------------------------------------------------------------------
extern "C" void kernel_launch(void* const* d_in, const int* in_sizes, int n_in,
                              void* d_out, int out_size, void* d_ws, size_t ws_size,
                              hipStream_t stream) {
    (void)in_sizes; (void)n_in; (void)out_size; (void)ws_size;
    const float* z   = (const float*)d_in[0];     // (32,256,32,32) f32
    const float* emb = (const float*)d_in[1];     // (1024,256) f32

    float* zq   = (float*)d_out;                  // 8388608 f32
    float* loss = (float*)d_out + (size_t)Np * Dd;            // 1 f32
    int*   idx  = (int*)((float*)d_out + (size_t)Np * Dd + 1);// 32768 i32

    uint32_t* ebf   = (uint32_t*)d_ws;                            // 512 KB bf16 codebook
    float* enorm    = (float*)((char*)d_ws + 512 * 1024);         // 4 KB norms
    float* partials = (float*)((char*)d_ws + 512 * 1024 + 4096);  // 128 KB partials

    vq_prep<<<Kc, 64, 0, stream>>>(emb, ebf, enorm);
    vq_argmin<<<Np / 128, 256, 0, stream>>>(z, ebf, enorm, idx);
    vq_gather<<<dim3(Np / 256, Cc), 256, 0, stream>>>(z, emb, idx, zq, partials);
    vq_loss_reduce<<<1, 256, 0, stream>>>(partials, loss);
}
